// GeneralAttention_41214506172673
// MI455X (gfx1250) — compile-verified, matrix operand order checked by reading, not run on hardware
//
#include <hip/hip_runtime.h>

typedef __bf16 bf16;
typedef __attribute__((ext_vector_type(16))) __bf16 v16bf;
typedef __attribute__((ext_vector_type(8)))  __bf16 v8bf;
typedef __attribute__((ext_vector_type(8)))  float  v8f;

static constexpr int Bsz = 2, S = 2048, D = 2048, H = 32, KVH = 8, HD = 64, REP = 4;
#define LOG2E 1.44269504088896340736f

static __device__ __forceinline__ v16bf make_frag(v8bf lo, v8bf hi) {
  return __builtin_shufflevector(lo, hi, 0,1,2,3,4,5,6,7,8,9,10,11,12,13,14,15);
}
static __device__ __forceinline__ v8f wmma_bf16(v16bf a, v16bf b, v8f c) {
  return __builtin_amdgcn_wmma_f32_16x16x32_bf16(false, a, false, b, (short)0, c, false, false);
}

// CDNA5 async global->LDS DMA (ASYNCcnt-tracked). Copies 16B per active lane.
// LDS byte address = low 32 bits of the flat shared-pointer.
static __device__ __forceinline__ void async_copy16(void* lds, const void* gsrc) {
  unsigned l = (unsigned)(uintptr_t)lds;
  asm volatile("global_load_async_to_lds_b128 %0, %1, off" :: "v"(l), "v"(gsrc) : "memory");
}
static __device__ __forceinline__ void wait_async0() {
  asm volatile("s_wait_asynccnt 0" ::: "memory");
}

// ---------------- fp32 -> bf16 (plain) ----------------
__global__ void cvt_f32_bf16(const float* __restrict__ in, bf16* __restrict__ out, int n) {
  int i = blockIdx.x * blockDim.x + threadIdx.x;
  int stride = gridDim.x * blockDim.x;
  for (; i < n; i += stride) out[i] = (bf16)in[i];
}

// ---------------- fp32 -> bf16 with transpose: in K x N  ->  out N x K ----------------
__global__ __launch_bounds__(256) void cvt_t_f32_bf16(const float* __restrict__ in,
                                                      bf16* __restrict__ out, int K, int N) {
  __shared__ float tile[32][33];
  const int k0 = blockIdx.y * 32, n0 = blockIdx.x * 32;
  const int tx = threadIdx.x & 31, ty = threadIdx.x >> 5;  // 32 x 8
#pragma unroll
  for (int i = 0; i < 32; i += 8)
    tile[ty + i][tx] = in[(size_t)(k0 + ty + i) * N + n0 + tx];
  __syncthreads();
#pragma unroll
  for (int i = 0; i < 32; i += 8)
    out[(size_t)(n0 + ty + i) * K + k0 + tx] = (bf16)tile[tx][ty + i];
}

// ---------------- WMMA GEMM: C[M,N] = A[M,K] * Wt[N,K]^T ----------------
// WG 256 threads, tile 64(M) x 128(N); 8 waves, each 32x32 (4 wmma per k-step).
// Async double-buffered LDS staging; Wt is pre-transposed so all copies are contiguous.
// MODE: 0 = bf16 row-major, 1 = f32 row-major, 2 = bf16 transposed [b,g,d,s] (for V).
template <int MODE>
__global__ __launch_bounds__(256) void gemm_bf16(const bf16* __restrict__ A,
                                                 const bf16* __restrict__ Wt,
                                                 void* __restrict__ Cptr,
                                                 int M, int N, int K) {
  __shared__ __align__(16) bf16 As[2][64][40];    // [m][k], 80B stride
  __shared__ __align__(16) bf16 Bs[2][128][40];   // [n][k], 80B stride
  const int tid = threadIdx.x;
  const int w = tid >> 5, lane = tid & 31;
  const int lm = lane & 15, kb = (lane >> 4) << 3, mb = (lane >> 4) << 3;
  const int m0 = blockIdx.y * 64, n0 = blockIdx.x * 128;
  const int mbase = (w >> 2) * 32, nc = (w & 3) * 32;
  v8f c00 = {}, c01 = {}, c10 = {}, c11 = {};

  auto loadAB = [&](int k0, int buf) {
    {  // A tile 64x32: 256 chunks of 16B
      int r = tid >> 2, c = (tid & 3) * 8;
      async_copy16(&As[buf][r][c], &A[(size_t)(m0 + r) * K + k0 + c]);
    }
#pragma unroll
    for (int i = 0; i < 2; ++i) {  // B tile 128x32 (from Wt rows): 512 chunks
      int idx = tid + i * 256;
      int n = idx >> 2, c = (idx & 3) * 8;
      async_copy16(&Bs[buf][n][c], &Wt[(size_t)(n0 + n) * K + k0 + c]);
    }
  };

  loadAB(0, 0);
  const int steps = K >> 5;
  for (int j = 0; j < steps; ++j) {
    const int buf = j & 1;
    wait_async0();
    __syncthreads();
    if (j + 1 < steps) loadAB((j + 1) << 5, buf ^ 1);

    v16bf a0 = make_frag(*(const v8bf*)&As[buf][mbase + lm][kb],
                         *(const v8bf*)&As[buf][mbase + lm][16 + kb]);
    v16bf a1 = make_frag(*(const v8bf*)&As[buf][mbase + 16 + lm][kb],
                         *(const v8bf*)&As[buf][mbase + 16 + lm][16 + kb]);
    v16bf b0 = make_frag(*(const v8bf*)&Bs[buf][nc + lm][kb],
                         *(const v8bf*)&Bs[buf][nc + lm][16 + kb]);
    v16bf b1 = make_frag(*(const v8bf*)&Bs[buf][nc + 16 + lm][kb],
                         *(const v8bf*)&Bs[buf][nc + 16 + lm][16 + kb]);
    c00 = wmma_bf16(a0, b0, c00);
    c01 = wmma_bf16(a0, b1, c01);
    c10 = wmma_bf16(a1, b0, c10);
    c11 = wmma_bf16(a1, b1, c11);
  }

  auto store_frag = [&](v8f c, int rowbase, int colbase) {
#pragma unroll
    for (int r = 0; r < 8; ++r) {
      int row = rowbase + r + mb;
      int col = colbase + lm;
      if constexpr (MODE == 1) {
        ((float*)Cptr)[(size_t)row * N + col] = c[r];
      } else if constexpr (MODE == 0) {
        ((bf16*)Cptr)[(size_t)row * N + col] = (bf16)c[r];
      } else {  // V: out[((b*KVH+g)*HD + d) * S + s], row = b*S+s, col = g*HD+d
        int bb = row >> 11, s = row & (S - 1);
        ((bf16*)Cptr)[((size_t)bb * (KVH * HD) + col) * S + s] = (bf16)c[r];
      }
    }
  };
  store_frag(c00, m0 + mbase, n0 + nc);
  store_frag(c01, m0 + mbase, n0 + nc + 16);
  store_frag(c10, m0 + mbase + 16, n0 + nc);
  store_frag(c11, m0 + mbase + 16, n0 + nc + 16);
}

// ---------------- Flash attention (causal, GQA) ----------------
// Grid: (S/128, H, B). WG 256 threads = 8 waves; wave w owns q rows [q0+16w, q0+16w+16).
// K/V blocks async double-buffered; V arrives pre-transposed [b,g,d,s].
__global__ __launch_bounds__(256) void flash_attn(const bf16* __restrict__ Q,
                                                  const bf16* __restrict__ Kg,
                                                  const bf16* __restrict__ Vt,
                                                  bf16* __restrict__ O) {
  __shared__ __align__(16) bf16 Ks[2][32][72];    // [key][dim], 144B stride
  __shared__ __align__(16) bf16 Vs[2][64][40];    // [dim][key], 80B stride
  __shared__ __align__(16) bf16 Ps[8][16][40];    // per-wave P tile 16x32
  const int tid = threadIdx.x, w = tid >> 5, lane = tid & 31;
  const int lm = lane & 15, kb = (lane >> 4) << 3, mb = (lane >> 4) << 3;
  const int q0 = blockIdx.x * 128;
  const int h = blockIdx.y, b = blockIdx.z, g = h / REP;
  const int qw = q0 + w * 16;

  const bf16* qrow = Q + ((size_t)(b * S + qw + lm) * H + h) * HD;
  v16bf qa0 = make_frag(*(const v8bf*)(qrow + kb),      *(const v8bf*)(qrow + 16 + kb));
  v16bf qa1 = make_frag(*(const v8bf*)(qrow + 32 + kb), *(const v8bf*)(qrow + 48 + kb));

  v8f o0 = {}, o1 = {}, o2 = {}, o3 = {};
  float mx[8], sum[8];
#pragma unroll
  for (int r = 0; r < 8; ++r) { mx[r] = -__builtin_inff(); sum[r] = 0.f; }

  auto loadKV = [&](int j, int buf) {
    {  // K block: 32 keys x 64 dims row-major = 256 chunks of 16B
      int key = tid >> 3, dc = (tid & 7) * 8;
      async_copy16(&Ks[buf][key][dc],
                   Kg + ((size_t)(b * S + j * 32 + key) * KVH + g) * HD + dc);
    }
    {  // V block from Vt: 64 dims x 32 keys = 256 chunks of 16B
      int d = tid >> 2, kc = (tid & 3) * 8;
      async_copy16(&Vs[buf][d][kc],
                   Vt + ((size_t)(b * KVH + g) * HD + d) * S + j * 32 + kc);
    }
  };

  const int jend = q0 / 32 + 4;  // causal bound (32-key blocks)
  loadKV(0, 0);
  for (int j = 0; j < jend; ++j) {
    const int buf = j & 1;
    wait_async0();
    __syncthreads();
    if (j + 1 < jend) loadKV(j + 1, buf ^ 1);

    if (j * 32 <= qw + 15) {  // wave-uniform causal skip
      v16bf k0lo = make_frag(*(const v8bf*)&Ks[buf][lm][kb],           *(const v8bf*)&Ks[buf][lm][16 + kb]);
      v16bf k0hi = make_frag(*(const v8bf*)&Ks[buf][lm][32 + kb],      *(const v8bf*)&Ks[buf][lm][48 + kb]);
      v16bf k1lo = make_frag(*(const v8bf*)&Ks[buf][16 + lm][kb],      *(const v8bf*)&Ks[buf][16 + lm][16 + kb]);
      v16bf k1hi = make_frag(*(const v8bf*)&Ks[buf][16 + lm][32 + kb], *(const v8bf*)&Ks[buf][16 + lm][48 + kb]);
      v8f s0 = {}, s1 = {};
      s0 = wmma_bf16(qa0, k0lo, s0); s0 = wmma_bf16(qa1, k0hi, s0);
      s1 = wmma_bf16(qa0, k1lo, s1); s1 = wmma_bf16(qa1, k1hi, s1);

      const float scale = 0.125f;  // 1/sqrt(HD)
#pragma unroll
      for (int r = 0; r < 8; ++r) {
        int qg = qw + r + mb;
        float v0 = s0[r] * scale, v1 = s1[r] * scale;
        if (j * 32 + lm > qg)      v0 = -3e38f;
        if (j * 32 + 16 + lm > qg) v1 = -3e38f;
        float rm = fmaxf(v0, v1);
        rm = fmaxf(rm, __shfl_xor(rm, 1));
        rm = fmaxf(rm, __shfl_xor(rm, 2));
        rm = fmaxf(rm, __shfl_xor(rm, 4));
        rm = fmaxf(rm, __shfl_xor(rm, 8));
        float nm = fmaxf(mx[r], rm);
        float corr = __builtin_amdgcn_exp2f((mx[r] - nm) * LOG2E);
        float p0 = __builtin_amdgcn_exp2f((v0 - nm) * LOG2E);
        float p1 = __builtin_amdgcn_exp2f((v1 - nm) * LOG2E);
        float ps = p0 + p1;
        ps += __shfl_xor(ps, 1); ps += __shfl_xor(ps, 2);
        ps += __shfl_xor(ps, 4); ps += __shfl_xor(ps, 8);
        sum[r] = sum[r] * corr + ps;
        mx[r] = nm;
        o0[r] *= corr; o1[r] *= corr; o2[r] *= corr; o3[r] *= corr;
        int m = r + mb;                  // C-layout -> per-wave LDS -> A-layout
        Ps[w][m][lm]      = (bf16)p0;
        Ps[w][m][16 + lm] = (bf16)p1;
      }
      v16bf pa  = make_frag(*(const v8bf*)&Ps[w][lm][kb],         *(const v8bf*)&Ps[w][lm][16 + kb]);
      v16bf vb0 = make_frag(*(const v8bf*)&Vs[buf][lm][kb],       *(const v8bf*)&Vs[buf][lm][16 + kb]);
      v16bf vb1 = make_frag(*(const v8bf*)&Vs[buf][16 + lm][kb],  *(const v8bf*)&Vs[buf][16 + lm][16 + kb]);
      v16bf vb2 = make_frag(*(const v8bf*)&Vs[buf][32 + lm][kb],  *(const v8bf*)&Vs[buf][32 + lm][16 + kb]);
      v16bf vb3 = make_frag(*(const v8bf*)&Vs[buf][48 + lm][kb],  *(const v8bf*)&Vs[buf][48 + lm][16 + kb]);
      o0 = wmma_bf16(pa, vb0, o0);
      o1 = wmma_bf16(pa, vb1, o1);
      o2 = wmma_bf16(pa, vb2, o2);
      o3 = wmma_bf16(pa, vb3, o3);
    }
  }

#pragma unroll
  for (int r = 0; r < 8; ++r) {
    float inv = 1.f / sum[r];
    size_t row = (size_t)(b * S + qw + r + mb);
    bf16* dst = O + (row * H + h) * HD;
    dst[lm]      = (bf16)(o0[r] * inv);
    dst[16 + lm] = (bf16)(o1[r] * inv);
    dst[32 + lm] = (bf16)(o2[r] * inv);
    dst[48 + lm] = (bf16)(o3[r] * inv);
  }
}

// ---------------- host launch ----------------
extern "C" void kernel_launch(void* const* d_in, const int* in_sizes, int n_in,
                              void* d_out, int out_size, void* d_ws, size_t ws_size,
                              hipStream_t stream) {
  (void)in_sizes; (void)n_in; (void)out_size; (void)ws_size;
  const float* x  = (const float*)d_in[0];
  const float* wq = (const float*)d_in[1];
  const float* wk = (const float*)d_in[2];
  const float* wv = (const float*)d_in[3];
  const float* wo = (const float*)d_in[4];
  // d_in[5] = causal mask: handled analytically.

  char* ws = (char*)d_ws;
  size_t off = 0;
  auto alloc = [&](size_t elems) { bf16* p = (bf16*)(ws + off); off += elems * sizeof(bf16); return p; };
  const size_t MD = (size_t)Bsz * S;  // 4096 rows
  bf16* xb   = alloc(MD * D);
  bf16* wqt  = alloc((size_t)D * H * HD);     // transposed: [n][k]
  bf16* wkt  = alloc((size_t)D * KVH * HD);
  bf16* wvt  = alloc((size_t)D * KVH * HD);
  bf16* wot  = alloc((size_t)H * HD * D);
  bf16* Qb   = alloc(MD * H * HD);
  bf16* Kb   = alloc(MD * KVH * HD);
  bf16* Vtb  = alloc(MD * KVH * HD);          // V pre-transposed [b,g,d,s]
  bf16* Ab   = alloc(MD * H * HD);

  // 1) conversions (weights transposed for async-friendly GEMM B tiles)
  cvt_f32_bf16<<<1024, 256, 0, stream>>>(x, xb, (int)(MD * D));
  cvt_t_f32_bf16<<<dim3((H * HD) / 32, D / 32),  256, 0, stream>>>(wq, wqt, D, H * HD);
  cvt_t_f32_bf16<<<dim3((KVH * HD) / 32, D / 32), 256, 0, stream>>>(wk, wkt, D, KVH * HD);
  cvt_t_f32_bf16<<<dim3((KVH * HD) / 32, D / 32), 256, 0, stream>>>(wv, wvt, D, KVH * HD);
  cvt_t_f32_bf16<<<dim3(D / 32, (H * HD) / 32),  256, 0, stream>>>(wo, wot, H * HD, D);

  // 2) QKV projections (WMMA bf16, async double-buffered)
  gemm_bf16<0><<<dim3((H * HD) / 128, MD / 64), 256, 0, stream>>>(xb, wqt, Qb, (int)MD, H * HD, D);
  gemm_bf16<0><<<dim3((KVH * HD) / 128, MD / 64), 256, 0, stream>>>(xb, wkt, Kb, (int)MD, KVH * HD, D);
  gemm_bf16<2><<<dim3((KVH * HD) / 128, MD / 64), 256, 0, stream>>>(xb, wvt, Vtb, (int)MD, KVH * HD, D);

  // 3) causal flash attention
  flash_attn<<<dim3(S / 128, H, Bsz), 256, 0, stream>>>(Qb, Kb, Vtb, Ab);

  // 4) output projection -> fp32 d_out
  gemm_bf16<1><<<dim3(D / 128, MD / 64), 256, 0, stream>>>(Ab, wot, d_out, (int)MD, D, H * HD);
}
